// DyGDecoder_86758339379618
// MI455X (gfx1250) — matrix operand order repaired and checked by reading, not compiled, not run on hardware
//
#include <hip/hip_runtime.h>
#include <hip/hip_bf16.h>
#include <math.h>

typedef __attribute__((ext_vector_type(16))) _Float16 v16h;
typedef __attribute__((ext_vector_type(8)))  _Float16 h8;
typedef __attribute__((ext_vector_type(8)))  float    v8f;

// ---------------------------------------------------------------------------
// helpers
// ---------------------------------------------------------------------------
__device__ __forceinline__ v16h pack16(h8 lo, h8 hi) {
  v16h a;
#pragma unroll
  for (int e = 0; e < 8; ++e) { a[e] = lo[e]; a[e + 8] = hi[e]; }
  return a;
}

__device__ __forceinline__ v8f wmma_f16(v16h a, v16h b, v8f c) {
  // D = A(16x32 f16) * B(32x16 f16) + C(16x16 f32)
  return __builtin_amdgcn_wmma_f32_16x16x32_f16(false, a, false, b,
                                                (short)0, c, false, false);
}

// ---------------------------------------------------------------------------
// weight transpose + fp32 -> fp16 :  Wt[n][k] = (f16) W[k][n]
// ---------------------------------------------------------------------------
__global__ __launch_bounds__(256) void wconv_k(const float* __restrict__ W,
                                               _Float16* __restrict__ Wt,
                                               int K, int N) {
  int idx = blockIdx.x * 256 + threadIdx.x;
  if (idx >= K * N) return;
  int n = idx / K, k = idx - n * K;
  Wt[idx] = (_Float16)W[(size_t)k * N + n];
}

// ---------------------------------------------------------------------------
// LN0 over concat(self, cross): one wave32 per row (D=256, 8 floats/lane)
// writes fp32 residual copy of raw input + fp16 normalized activations
// ---------------------------------------------------------------------------
__global__ __launch_bounds__(256) void ln0_k(const float* __restrict__ selfSeq,
                                             const float* __restrict__ crossSeq,
                                             const float* __restrict__ g,
                                             const float* __restrict__ bta,
                                             float* __restrict__ resid0,
                                             _Float16* __restrict__ xh) {
  int row  = blockIdx.x * 8 + (threadIdx.x >> 5);
  int lane = threadIdx.x & 31;
  int b = row >> 12, t = row & 4095;
  const float* src = (t < 2048)
      ? selfSeq  + ((size_t)(b * 2048 + t)) * 256
      : crossSeq + ((size_t)(b * 2048 + (t - 2048))) * 256;
  float x[8];
  *(float4*)&x[0] = *(const float4*)(src + lane * 8);
  *(float4*)&x[4] = *(const float4*)(src + lane * 8 + 4);
  float s = 0.f, sq = 0.f;
#pragma unroll
  for (int i = 0; i < 8; ++i) { s += x[i]; sq += x[i] * x[i]; }
#pragma unroll
  for (int o = 16; o; o >>= 1) {
    s  += __shfl_xor(s,  o, 32);
    sq += __shfl_xor(sq, o, 32);
  }
  float mean = s * (1.f / 256.f);
  float var  = sq * (1.f / 256.f) - mean * mean;
  float rstd = rsqrtf(var + 1e-5f);
#pragma unroll
  for (int i = 0; i < 8; ++i) {
    int col = lane * 8 + i;
    resid0[(size_t)row * 256 + col] = x[i];
    xh[(size_t)row * 256 + col] =
        (_Float16)((x[i] - mean) * rstd * g[col] + bta[col]);
  }
}

// LN1: input = fp32 residual buffer, output fp16 only
__global__ __launch_bounds__(256) void ln1_k(const float* __restrict__ src,
                                             const float* __restrict__ g,
                                             const float* __restrict__ bta,
                                             _Float16* __restrict__ xh) {
  int row  = blockIdx.x * 8 + (threadIdx.x >> 5);
  int lane = threadIdx.x & 31;
  const float* sp = src + (size_t)row * 256;
  float x[8];
  *(float4*)&x[0] = *(const float4*)(sp + lane * 8);
  *(float4*)&x[4] = *(const float4*)(sp + lane * 8 + 4);
  float s = 0.f, sq = 0.f;
#pragma unroll
  for (int i = 0; i < 8; ++i) { s += x[i]; sq += x[i] * x[i]; }
#pragma unroll
  for (int o = 16; o; o >>= 1) {
    s  += __shfl_xor(s,  o, 32);
    sq += __shfl_xor(sq, o, 32);
  }
  float mean = s * (1.f / 256.f);
  float var  = sq * (1.f / 256.f) - mean * mean;
  float rstd = rsqrtf(var + 1e-5f);
#pragma unroll
  for (int i = 0; i < 8; ++i) {
    int col = lane * 8 + i;
    xh[(size_t)row * 256 + col] =
        (_Float16)((x[i] - mean) * rstd * g[col] + bta[col]);
  }
}

// ---------------------------------------------------------------------------
// Generic NT WMMA GEMM:  C[M][N] = A[M][K] * Wt[N][K]^T   (f16 in, f32 acc)
// Block = 256 threads (8 waves), tile 128(M) x 64(N) x 32(K).
// Each wave owns a 32x32 quadrant = 4 WMMA 16x16 tiles.
// MODE 0: QKV (region-select weights, scatter to Q/K/V^T head buffers)
// MODE 1: proj   -> f32 resid2 = resid0 + bias + C
// MODE 2: ff1    -> fp16 gelu(C + bias)
// MODE 3: ff2    -> d_out = resid2 + bias + C  (with self/cross split)
// ---------------------------------------------------------------------------
template <int MODE>
__global__ __launch_bounds__(256) void gemm_k(
    const _Float16* __restrict__ A, int K,
    const _Float16* __restrict__ W0, const _Float16* __restrict__ W1,
    const float* __restrict__ bias0, const float* __restrict__ bias1,
    const float* __restrict__ resid,
    _Float16* __restrict__ outQ, _Float16* __restrict__ outK,
    _Float16* __restrict__ outV, _Float16* __restrict__ outH,
    float* __restrict__ outF) {
  __shared__ _Float16 As[128][32];
  __shared__ _Float16 Bs[64][32];
  const int tid  = threadIdx.x;
  const int row0 = blockIdx.x * 128;
  const int n0   = blockIdx.y * 64;
  const bool crossRegion = ((row0 >> 11) & 1) != 0;  // 2048-row granularity
  const _Float16* W   = (MODE == 0 && crossRegion) ? W1 : W0;
  const float*    bia = (MODE == 0 && crossRegion) ? bias1 : bias0;
  const int wave = tid >> 5, lane = tid & 31;
  const int lh = lane >> 4, l15 = lane & 15;
  const int wm = (wave >> 1) * 32, wn = (wave & 1) * 32;
  v8f c00 = {}, c01 = {}, c10 = {}, c11 = {};

  for (int k0 = 0; k0 < K; k0 += 32) {
    __syncthreads();
    {  // A tile: 128x32 halves, 16 halves per thread
      int r = tid >> 1, seg = (tid & 1) * 16;
      const uint4* s = (const uint4*)(A + (size_t)(row0 + r) * K + (k0 + seg));
      *(uint4*)&As[r][seg]     = s[0];
      *(uint4*)&As[r][seg + 8] = s[1];
    }
    {  // B tile: 64x32 halves (already [n][k]), 8 halves per thread
      int n = tid >> 2, seg = (tid & 3) * 8;
      *(uint4*)&Bs[n][seg] =
          *(const uint4*)(W + (size_t)(n0 + n) * K + (k0 + seg));
    }
    __syncthreads();
    // B fragments: lane holds n=l15, k = lh*8+(e&7)+(e>=8?16:0)
    v16h b0 = pack16(*(const h8*)&Bs[wn + l15][lh * 8],
                     *(const h8*)&Bs[wn + l15][lh * 8 + 16]);
    v16h b1 = pack16(*(const h8*)&Bs[wn + 16 + l15][lh * 8],
                     *(const h8*)&Bs[wn + 16 + l15][lh * 8 + 16]);
    // A fragments: lane holds m=l15, k = lh*8+(e&7)+(e>=8?16:0)
    v16h a0 = pack16(*(const h8*)&As[wm + l15][lh * 8],
                     *(const h8*)&As[wm + l15][lh * 8 + 16]);
    v16h a1 = pack16(*(const h8*)&As[wm + 16 + l15][lh * 8],
                     *(const h8*)&As[wm + 16 + l15][lh * 8 + 16]);
    c00 = wmma_f16(a0, b0, c00);
    c01 = wmma_f16(a0, b1, c01);
    c10 = wmma_f16(a1, b0, c10);
    c11 = wmma_f16(a1, b1, c11);
  }

  // epilogue: C layout -> lane column n = l15, rows m = lh*8 + r
#pragma unroll
  for (int i = 0; i < 2; ++i) {
#pragma unroll
    for (int j = 0; j < 2; ++j) {
      v8f cc = (i == 0) ? (j == 0 ? c00 : c01) : (j == 0 ? c10 : c11);
      int ncol = n0 + wn + j * 16 + l15;
#pragma unroll
      for (int r = 0; r < 8; ++r) {
        int mrow = row0 + wm + i * 16 + lh * 8 + r;
        float v = cc[r];
        if (MODE == 0) {
          v += bia[ncol];
          int t = mrow & 4095, b = mrow >> 12;
          _Float16 hv = (_Float16)v;
          if (ncol < 256) {
            int h = ncol >> 6, d = ncol & 63;
            outQ[((size_t)((b * 4 + h) * 4096 + t)) * 64 + d] = hv;
          } else if (ncol < 512) {
            int h = (ncol - 256) >> 6, d = ncol & 63;
            outK[((size_t)((b * 4 + h) * 4096 + t)) * 64 + d] = hv;
          } else {
            int h = (ncol - 512) >> 6, d = ncol & 63;
            outV[((size_t)((b * 4 + h) * 64 + d)) * 4096 + t] = hv;  // V^T
          }
        } else if (MODE == 1) {
          outF[(size_t)mrow * 256 + ncol] =
              resid[(size_t)mrow * 256 + ncol] + bia[ncol] + v;
        } else if (MODE == 2) {
          float x = v + bia[ncol];
          float ge = 0.5f * x * (1.f + erff(x * 0.70710678118654752f));
          outH[(size_t)mrow * 1024 + ncol] = (_Float16)ge;
        } else {  // MODE 3: final output with (self, cross) split
          float x = resid[(size_t)mrow * 256 + ncol] + bia[ncol] + v;
          int t = mrow & 4095, b = mrow >> 12;
          size_t dst = (t < 2048)
              ? ((size_t)(b * 2048 + t)) * 256 + ncol
              : (size_t)2 * 2048 * 256 +
                    ((size_t)(b * 2048 + (t - 2048))) * 256 + ncol;
          outF[dst] = x;
        }
      }
    }
  }
}

// ---------------------------------------------------------------------------
// Flash attention, transposed score tiles.
// One wave per 16-query tile; block = 8 waves = 128 queries for one (b,h).
// Keys streamed in 32-chunks: 4 WMMAs S^T = K*Q^T, online softmax across
// the (lane, lane^16) pair, P via LDS re-fragmentation, 4 WMMAs O^T += V^T*P.
// ---------------------------------------------------------------------------
__global__ __launch_bounds__(256) void attn_k(
    const _Float16* __restrict__ Qb, const _Float16* __restrict__ Kb,
    const _Float16* __restrict__ Vtb,
    const float* __restrict__ tSelf, const float* __restrict__ tCross,
    _Float16* __restrict__ ybuf) {
  __shared__ float tls[4096];
  __shared__ _Float16 Pbuf[8][16][32];
  const int bh = blockIdx.y, b = bh >> 2, h = bh & 3;
  for (int i = threadIdx.x; i < 4096; i += 256)
    tls[i] = (i < 2048) ? tSelf[b * 2048 + i] : tCross[b * 2048 + (i - 2048)];
  __syncthreads();

  const int wave = threadIdx.x >> 5, lane = threadIdx.x & 31;
  const int lh = lane >> 4, l15 = lane & 15;
  const int q0 = blockIdx.x * 128 + wave * 16;
  const size_t bh64 = (size_t)(b * 4 + h);

  // Q as B-fragments (held for whole key loop): n=query, k=d table order
  const _Float16* Qrow = Qb + (bh64 * 4096 + (size_t)(q0 + l15)) * 64;
  v16h bq0 = pack16(*(const h8*)(Qrow + lh * 8),
                    *(const h8*)(Qrow + lh * 8 + 16));
  v16h bq1 = pack16(*(const h8*)(Qrow + 32 + lh * 8),
                    *(const h8*)(Qrow + 32 + lh * 8 + 16));
  const float tq = tls[q0 + l15];
  const _Float16* Kbase = Kb + bh64 * 4096 * 64;
  const _Float16* Vbase = Vtb + bh64 * 64 * 4096;

  float mrun = -1e9f, lrun = 0.f;
  v8f o0 = {}, o1 = {}, o2 = {}, o3 = {};  // O^T[d][q], d = dt*16 + lh*8 + r

  for (int k0 = 0; k0 < 4096; k0 += 32) {
    v8f s0 = {}, s1 = {};
    {  // keys k0..k0+15
      const _Float16* kr = Kbase + (size_t)(k0 + l15) * 64;
      v16h a0 = pack16(*(const h8*)(kr + lh * 8),
                       *(const h8*)(kr + lh * 8 + 16));
      v16h a1 = pack16(*(const h8*)(kr + 32 + lh * 8),
                       *(const h8*)(kr + 32 + lh * 8 + 16));
      s0 = wmma_f16(a0, bq0, s0);
      s0 = wmma_f16(a1, bq1, s0);
    }
    {  // keys k0+16..k0+31
      const _Float16* kr = Kbase + (size_t)(k0 + 16 + l15) * 64;
      v16h a0 = pack16(*(const h8*)(kr + lh * 8),
                       *(const h8*)(kr + lh * 8 + 16));
      v16h a1 = pack16(*(const h8*)(kr + 32 + lh * 8),
                       *(const h8*)(kr + 32 + lh * 8 + 16));
      s1 = wmma_f16(a0, bq0, s1);
      s1 = wmma_f16(a1, bq1, s1);
    }
    // scale + time mask (S^T tile: lane column = query l15, rows = keys)
    float sc[16];
#pragma unroll
    for (int r = 0; r < 8; ++r) {
      float v0 = s0[r] * 0.125f;
      if (!(tq >= tls[k0 + lh * 8 + r])) v0 = -1e9f;
      sc[r] = v0;
      float v1 = s1[r] * 0.125f;
      if (!(tq >= tls[k0 + 16 + lh * 8 + r])) v1 = -1e9f;
      sc[8 + r] = v1;
    }
    // online softmax: pair (lane, lane^16) covers all 32 keys of this query
    float cm = sc[0];
#pragma unroll
    for (int i = 1; i < 16; ++i) cm = fmaxf(cm, sc[i]);
    cm = fmaxf(cm, __shfl_xor(cm, 16, 32));
    float mnew = fmaxf(mrun, cm);
    float ps = 0.f;
    _Float16 ph[16];
#pragma unroll
    for (int i = 0; i < 16; ++i) {
      float p = __expf(sc[i] - mnew);
      ps += p;
      ph[i] = (_Float16)p;
    }
    ps += __shfl_xor(ps, 16, 32);
    float scl = __expf(mrun - mnew);
    lrun = lrun * scl + ps;
    mrun = mnew;
#pragma unroll
    for (int r = 0; r < 8; ++r) {
      o0[r] *= scl; o1[r] *= scl; o2[r] *= scl; o3[r] *= scl;
    }
    // stage P in LDS as [q][key] then reload in B-fragment layout
    {
      h8 plo, phi;
#pragma unroll
      for (int e = 0; e < 8; ++e) { plo[e] = ph[e]; phi[e] = ph[8 + e]; }
      *(h8*)&Pbuf[wave][l15][lh * 8]      = plo;
      *(h8*)&Pbuf[wave][l15][16 + lh * 8] = phi;
    }
    v16h pf = pack16(*(const h8*)&Pbuf[wave][l15][lh * 8],
                     *(const h8*)&Pbuf[wave][l15][lh * 8 + 16]);
    // O^T += V^T * P  (4 d-tiles of 16)
    {
      const _Float16* vr = Vbase + (size_t)(l15) * 4096 + k0;
      o0 = wmma_f16(pack16(*(const h8*)(vr + lh * 8),
                           *(const h8*)(vr + lh * 8 + 16)), pf, o0);
    }
    {
      const _Float16* vr = Vbase + (size_t)(16 + l15) * 4096 + k0;
      o1 = wmma_f16(pack16(*(const h8*)(vr + lh * 8),
                           *(const h8*)(vr + lh * 8 + 16)), pf, o1);
    }
    {
      const _Float16* vr = Vbase + (size_t)(32 + l15) * 4096 + k0;
      o2 = wmma_f16(pack16(*(const h8*)(vr + lh * 8),
                           *(const h8*)(vr + lh * 8 + 16)), pf, o2);
    }
    {
      const _Float16* vr = Vbase + (size_t)(48 + l15) * 4096 + k0;
      o3 = wmma_f16(pack16(*(const h8*)(vr + lh * 8),
                           *(const h8*)(vr + lh * 8 + 16)), pf, o3);
    }
  }

  float inv = 1.f / lrun;
  int token = b * 4096 + q0 + l15;
  _Float16* yr = ybuf + (size_t)token * 256 + h * 64;
#pragma unroll
  for (int r = 0; r < 8; ++r) {
    int dbase = lh * 8 + r;
    yr[dbase]      = (_Float16)(o0[r] * inv);
    yr[16 + dbase] = (_Float16)(o1[r] * inv);
    yr[32 + dbase] = (_Float16)(o2[r] * inv);
    yr[48 + dbase] = (_Float16)(o3[r] * inv);
  }
}

// ---------------------------------------------------------------------------
// launch
// ---------------------------------------------------------------------------
extern "C" void kernel_launch(void* const* d_in, const int* in_sizes, int n_in,
                              void* d_out, int out_size, void* d_ws,
                              size_t ws_size, hipStream_t stream) {
  (void)in_sizes; (void)n_in; (void)out_size; (void)ws_size;
  const float* self_seq = (const float*)d_in[0];
  const float* cross_seq = (const float*)d_in[1];
  const float* t_self = (const float*)d_in[2];
  const float* t_cross = (const float*)d_in[3];
  const float* ln0_g = (const float*)d_in[4];
  const float* ln0_b = (const float*)d_in[5];
  const float* w_self = (const float*)d_in[6];
  const float* b_self = (const float*)d_in[7];
  const float* w_cross = (const float*)d_in[8];
  const float* b_cross = (const float*)d_in[9];
  const float* w_proj = (const float*)d_in[10];
  const float* b_proj = (const float*)d_in[11];
  const float* ln1_g = (const float*)d_in[12];
  const float* ln1_b = (const float*)d_in[13];
  const float* w_ff1 = (const float*)d_in[14];
  const float* b_ff1 = (const float*)d_in[15];
  const float* w_ff2 = (const float*)d_in[16];
  const float* b_ff2 = (const float*)d_in[17];

  char* ws = (char*)d_ws;
  size_t off = 0;
  auto alloc = [&](size_t bytes) -> char* {
    char* p = ws + off;
    off += (bytes + 255) & ~(size_t)255;
    return p;
  };
  _Float16* wtSelf  = (_Float16*)alloc((size_t)768 * 256 * 2);
  _Float16* wtCross = (_Float16*)alloc((size_t)768 * 256 * 2);
  _Float16* wtProj  = (_Float16*)alloc((size_t)256 * 256 * 2);
  _Float16* wtFf1   = (_Float16*)alloc((size_t)1024 * 256 * 2);
  _Float16* wtFf2   = (_Float16*)alloc((size_t)256 * 1024 * 2);
  _Float16* xh      = (_Float16*)alloc((size_t)8192 * 256 * 2);
  float*    resid0  = (float*)   alloc((size_t)8192 * 256 * 4);
  _Float16* Qb      = (_Float16*)alloc((size_t)8 * 4096 * 64 * 2);
  _Float16* Kb      = (_Float16*)alloc((size_t)8 * 4096 * 64 * 2);
  _Float16* Vtb     = (_Float16*)alloc((size_t)8 * 4096 * 64 * 2);
  _Float16* ybuf    = (_Float16*)alloc((size_t)8192 * 256 * 2);
  float*    resid2  = (float*)   alloc((size_t)8192 * 256 * 4);
  _Float16* ln1h    = (_Float16*)alloc((size_t)8192 * 256 * 2);
  _Float16* hbuf    = (_Float16*)alloc((size_t)8192 * 1024 * 2);

  // weights -> f16 transposed [N][K]
  wconv_k<<<(768 * 256 + 255) / 256, 256, 0, stream>>>(w_self, wtSelf, 256, 768);
  wconv_k<<<(768 * 256 + 255) / 256, 256, 0, stream>>>(w_cross, wtCross, 256, 768);
  wconv_k<<<(256 * 256 + 255) / 256, 256, 0, stream>>>(w_proj, wtProj, 256, 256);
  wconv_k<<<(1024 * 256 + 255) / 256, 256, 0, stream>>>(w_ff1, wtFf1, 256, 1024);
  wconv_k<<<(1024 * 256 + 255) / 256, 256, 0, stream>>>(w_ff2, wtFf2, 1024, 256);

  // LN0
  ln0_k<<<1024, 256, 0, stream>>>(self_seq, cross_seq, ln0_g, ln0_b, resid0, xh);
  // QKV projection (region-dependent weights)
  gemm_k<0><<<dim3(64, 12), 256, 0, stream>>>(xh, 256, wtSelf, wtCross, b_self,
                                              b_cross, nullptr, Qb, Kb, Vtb,
                                              nullptr, nullptr);
  // attention
  attn_k<<<dim3(32, 8), 256, 0, stream>>>(Qb, Kb, Vtb, t_self, t_cross, ybuf);
  // output proj + residual
  gemm_k<1><<<dim3(64, 4), 256, 0, stream>>>(ybuf, 256, wtProj, nullptr, b_proj,
                                             nullptr, resid0, nullptr, nullptr,
                                             nullptr, nullptr, resid2);
  // LN1
  ln1_k<<<1024, 256, 0, stream>>>(resid2, ln1_g, ln1_b, ln1h);
  // FF1 + GELU
  gemm_k<2><<<dim3(64, 16), 256, 0, stream>>>(ln1h, 256, wtFf1, nullptr, b_ff1,
                                              nullptr, nullptr, nullptr,
                                              nullptr, nullptr, hbuf, nullptr);
  // FF2 + residual + split output
  gemm_k<3><<<dim3(64, 4), 256, 0, stream>>>(hbuf, 1024, wtFf2, nullptr, b_ff2,
                                             nullptr, resid2, nullptr, nullptr,
                                             nullptr, nullptr, (float*)d_out);
}
